// Quantize_7602092114376
// MI455X (gfx1250) — matrix operand order repaired and checked by reading, not compiled, not run on hardware
//
#include <hip/hip_runtime.h>
#include <hip/hip_bf16.h>
#include <stdint.h>

typedef __attribute__((ext_vector_type(16))) _Float16 v16h;
typedef __attribute__((ext_vector_type(8)))  float    v8f;
typedef __attribute__((ext_vector_type(4)))  unsigned v4u;
typedef __attribute__((ext_vector_type(8)))  int      v8i;
typedef __attribute__((ext_vector_type(4)))  int      v4i;

#define NPIX   131072      // 32*64*64 pixels
#define CCH    64          // channels
#define KCODES 512         // codebook entries
#define HWP    4096        // 64*64 plane
#define CHW    262144      // 64*4096
#define QELEMS 8388608     // 32*64*64*64

// workspace layout (bytes)
#define WS_WF16_OFF 0          // 512*64 f16 = 65536 B
#define WS_WSQ_OFF  65536      // 512 f32   = 2048 B
#define WS_PART_OFF 67584      // 8192 f32  = 32768 B
#define NPART 8192

#ifndef USE_TDM
#define USE_TDM 1
#endif

// ---------------- prep: f32 codebook -> f16 + |w|^2 ----------------
__global__ void vq_prep(const float* __restrict__ W, _Float16* __restrict__ wf16,
                        float* __restrict__ wsq) {
  int k = blockIdx.x * blockDim.x + threadIdx.x;
  if (k >= KCODES) return;
  float s = 0.f;
#pragma unroll 8
  for (int c = 0; c < CCH; ++c) {
    float v = W[k * CCH + c];
    s += v * v;
    wf16[k * CCH + c] = (_Float16)v;
  }
  wsq[k] = s;
}

// ---------------- cross-lane helpers (ds_swizzle, group-of-32) ----------------
template <int PAT>
__device__ __forceinline__ void argmin_step(float (&best)[8], int (&bidx)[8]) {
#pragma unroll
  for (int r = 0; r < 8; ++r) {
    float ob = __builtin_bit_cast(
        float, __builtin_amdgcn_ds_swizzle(__builtin_bit_cast(int, best[r]), PAT));
    int oi = __builtin_amdgcn_ds_swizzle(bidx[r], PAT);
    bool take = (ob < best[r]) || (ob == best[r] && oi < bidx[r]);  // argmin tie: lowest idx
    best[r] = take ? ob : best[r];
    bidx[r] = take ? oi : bidx[r];
  }
}

template <int PAT>
__device__ __forceinline__ float red_step(float v) {
  return v + __builtin_bit_cast(float,
             __builtin_amdgcn_ds_swizzle(__builtin_bit_cast(int, v), PAT));
}

// ---------------- main: WMMA distances + argmin + gather ----------------
__launch_bounds__(256, 1)
__global__ void vq_main(const float* __restrict__ x, const float* __restrict__ W32,
                        const _Float16* __restrict__ wf16, const float* __restrict__ wsq,
                        float* __restrict__ out, float* __restrict__ partials) {
  __shared__ __align__(16) _Float16 lds_cb[KCODES * CCH];  // 65536 B, exactly the codebook

  const int tid  = (int)threadIdx.x;
  const int lane = tid & 31;
  const int wid  = tid >> 5;       // 8 waves
  const int g    = lane >> 4;      // K-half group (A layout) / channel-half (epilogue)
  const int m    = lane & 15;      // row (pixel) within tile / B column lane

  // ---- stage f16 codebook into LDS ----
#if USE_TDM && __has_builtin(__builtin_amdgcn_tensor_load_to_lds)
  if (__builtin_amdgcn_readfirstlane(wid) == 0) {   // uniform branch: one wave issues TDM
    const unsigned long long ga = (unsigned long long)(uintptr_t)wf16;
    const unsigned ldsoff = (unsigned)(uintptr_t)(&lds_cb[0]);
    // D# group0: count=1 | lds_addr | global_addr[56:0] | type=2
    v4u g0 = {1u, ldsoff, (unsigned)ga,
              (unsigned)((ga >> 32) & 0x1FFFFFFull) | (2u << 30)};
    // D# group1: data_size=1 (2B); tensor_dim0=32768; tensor_dim1=1;
    //            tile_dim0=32768; tile_dim1=1; dim0_stride=32768
    v8i g1 = {(int)(1u << 16), (int)0x80000000u, 0x00010000, (int)0x80000000u,
              1, 32768, 0, 0};
    v4i gz = {0, 0, 0, 0};
#if defined(__clang_major__) && (__clang_major__ >= 23)
    v8i gz8 = {0, 0, 0, 0, 0, 0, 0, 0};
    __builtin_amdgcn_tensor_load_to_lds(g0, g1, gz, gz, gz8, 0);
#else
    __builtin_amdgcn_tensor_load_to_lds(g0, g1, gz, gz, 0);
#endif
    __builtin_amdgcn_s_wait_tensorcnt(0);
  }
#else
  {
    const uint4* src = (const uint4*)wf16;
    uint4* dst = (uint4*)lds_cb;
#pragma unroll
    for (int i = 0; i < 16; ++i) dst[tid + i * 256] = src[tid + i * 256];
  }
#endif
  __syncthreads();

  // ---- per-wave 16-pixel tile; tiles never cross an image plane (4096 % 128 == 0) ----
  const int p0 = (int)blockIdx.x * 128 + wid * 16;
  const int n  = p0 >> 12;
  const int ip = p0 & 4095;
  const long xb = (long)n * CHW + ip + m;  // x[n][c][pixel m], add c*HWP

  // A fragments (-2*x, f16), contiguous k order (B uses the same):
  // half j of lane (g,m): K = 32*s + 16*g + j
  v16h a0, a1;
#pragma unroll
  for (int j = 0; j < 16; ++j) {
    const int ck = 16 * g + j;
    a0[j] = (_Float16)(-2.f * x[xb + (long)ck * HWP]);
    a1[j] = (_Float16)(-2.f * x[xb + (long)(ck + 32) * HWP]);
  }

  float best[8];
  int   bidx[8];
#pragma unroll
  for (int r = 0; r < 8; ++r) { best[r] = 3.4e38f; bidx[r] = 0; }

  // ---- 32 N-tiles of 16 codes; score = -2*x.w + |w|^2 (|x|^2 is row-constant) ----
#pragma unroll 4
  for (int t = 0; t < 32; ++t) {
    const int code = t * 16 + m;                 // this lane's B column code
    const float wq = wsq[code];                  // L2-hot global read
    const _Float16* bp = &lds_cb[code * CCH + 16 * g];
    v16h b0 = *(const v16h*)bp;                  // 2x ds_load_b128
    v16h b1 = *(const v16h*)(bp + 32);
    v8f acc;
#pragma unroll
    for (int r = 0; r < 8; ++r) acc[r] = wq;     // C init = |w|^2
    acc = __builtin_amdgcn_wmma_f32_16x16x32_f16(false, a0, false, b0,
                                                 (short)0, acc, false, false);
    acc = __builtin_amdgcn_wmma_f32_16x16x32_f16(false, a1, false, b1,
                                                 (short)0, acc, false, false);
#pragma unroll
    for (int r = 0; r < 8; ++r) {
      bool take = acc[r] < best[r];              // strict <: earliest (smallest) code wins
      best[r] = take ? acc[r] : best[r];
      bidx[r] = take ? code   : bidx[r];
    }
  }

  // ---- argmin across the 16 column lanes (keep g halves separate) ----
  argmin_step<0x1F | (1 << 10)>(best, bidx);
  argmin_step<0x1F | (2 << 10)>(best, bidx);
  argmin_step<0x1F | (4 << 10)>(best, bidx);
  argmin_step<0x1F | (8 << 10)>(best, bidx);

  // ---- route row results to owning lanes: row M lives at lane ((M&8)<<1)|(M&7), slot M&7 ----
  int sel = bidx[0];
#pragma unroll
  for (int r = 1; r < 8; ++r) sel = ((lane & 7) == r) ? bidx[r] : sel;
  const int srclane = ((m & 8) << 1) | (m & 7);
  const int idx_m = __builtin_amdgcn_ds_bpermute(srclane << 2, sel);

  // embed_idx output (one lane-half writes)
  if (!(lane & 16)) out[QELEMS + p0 + m] = (float)idx_m;

  // ---- fp32 gather + quantize scatter + loss partial (each half-wave does 32 channels) ----
  const float* wrow = W32 + idx_m * CCH;
  float lsum = 0.f;
#pragma unroll 8
  for (int j = 0; j < 32; ++j) {
    const int c = g * 32 + j;
    const float q  = wrow[c];
    const float xv = x[xb + (long)c * HWP];
    const float d  = xv - q;
    lsum += d * d;
    out[xb + (long)c * HWP] = q;                 // quantize, NCHW
  }
  lsum = red_step<0x1F | (1  << 10)>(lsum);
  lsum = red_step<0x1F | (2  << 10)>(lsum);
  lsum = red_step<0x1F | (4  << 10)>(lsum);
  lsum = red_step<0x1F | (8  << 10)>(lsum);
  lsum = red_step<0x1F | (16 << 10)>(lsum);
  if (lane == 0) partials[(int)blockIdx.x * 8 + wid] = lsum;  // deterministic slot
}

// ---------------- finish: fixed-order reduction of loss partials ----------------
__global__ void vq_finish(const float* __restrict__ partials, float* __restrict__ out) {
  __shared__ float red[256];
  float s = 0.f;
  for (int i = (int)threadIdx.x; i < NPART; i += 256) s += partials[i];
  red[threadIdx.x] = s;
  __syncthreads();
  for (int off = 128; off > 0; off >>= 1) {
    if ((int)threadIdx.x < off) red[threadIdx.x] += red[threadIdx.x + off];
    __syncthreads();
  }
  if (threadIdx.x == 0) {
    float mean = red[0] / (float)QELEMS;
    out[QELEMS + NPIX + 0] = mean;   // embed_loss
    out[QELEMS + NPIX + 1] = mean;   // commit_loss (identical in forward)
  }
}

extern "C" void kernel_launch(void* const* d_in, const int* in_sizes, int n_in,
                              void* d_out, int out_size, void* d_ws, size_t ws_size,
                              hipStream_t stream) {
  (void)in_sizes; (void)n_in; (void)out_size; (void)ws_size;
  const float* x  = (const float*)d_in[0];
  const float* W  = (const float*)d_in[1];
  _Float16* wf16  = (_Float16*)((char*)d_ws + WS_WF16_OFF);
  float* wsq      = (float*)((char*)d_ws + WS_WSQ_OFF);
  float* parts    = (float*)((char*)d_ws + WS_PART_OFF);
  float* out      = (float*)d_out;

  vq_prep<<<2, 256, 0, stream>>>(W, wf16, wsq);
  vq_main<<<1024, 256, 0, stream>>>(x, W, wf16, wsq, out, parts);
  vq_finish<<<1, 256, 0, stream>>>(parts, out);
}
